// Attention_16183436771801
// MI455X (gfx1250) — matrix operand order repaired and checked by reading, not compiled, not run on hardware
//
#include <hip/hip_runtime.h>
#include <hip/hip_bf16.h>

typedef _Float16 half_t;
typedef __attribute__((ext_vector_type(16))) _Float16 v16h;
typedef __attribute__((ext_vector_type(8)))  float    v8f;

#define NTOK   294          // tokens per window (6*7*7)
#define NWIN   256          // windows (16*16)
#define MTOT   75264        // NWIN * NTOK
#define DIM    256
#define HEADS  8
#define DHEAD  32
#define QK_SCALE 0.17677669529663687f   // 32^-0.5
#define NEG_BIG (-3.0e38f)

static __device__ __forceinline__ int imin(int a, int b) { return a < b ? a : b; }

// A-fragment K index for 16-bit 16x32 A tile (contiguous: k = 16*grp + h, matches B)
static __device__ __forceinline__ int ka_idx(int kb, int h) {
    return kb + h;
}

static __device__ __forceinline__ v8f wmma_f16(v16h a, v16h b, v8f c) {
    return __builtin_amdgcn_wmma_f32_16x16x32_f16(false, a, false, b, (short)0, c, false, false);
}

// ---------------------------------------------------------------------------
// Kernel 1: QKV projection. qkv[r][c] = sum_d xw[r][d] * w_qkv[c][d]
// grid (1176, 12), block 256. Block tile: 64 rows x 64 cols, K=256 in LDS.
// Epilogue scatters f16 into q/k/v workspaces laid out [win][head][tok][32].
// ---------------------------------------------------------------------------
__global__ __launch_bounds__(256) void qkv_kernel(const float* __restrict__ x,
                                                  const float* __restrict__ wqkv,
                                                  half_t* __restrict__ qw,
                                                  half_t* __restrict__ kw,
                                                  half_t* __restrict__ vw) {
    __shared__ alignas(32) half_t a_s[64 * 256];
    __shared__ alignas(32) half_t b_s[64 * 256];
    const int tid = threadIdx.x;
    const int m0 = blockIdx.x * 64;
    const int c0 = blockIdx.y * 64;

    // cooperative load: 4 threads per row, 64 f32->f16 each
    {
        const int rl  = tid >> 2;
        const int seg = (tid & 3) * 64;
        const int rg  = m0 + rl;
        const int b   = rg / NTOK;
        const int n   = rg - b * NTOK;
        const int l   = n / 49;
        const int s   = n - l * 49;
        // x flat layout: ((l*256 + win)*49 + s)*256 + d
        const float* asrc = x + (size_t)((l * 256 + b) * 49 + s) * 256 + seg;
        half_t* adst = a_s + rl * 256 + seg;
        const float* bsrc = wqkv + (size_t)(c0 + rl) * 256 + seg;
        half_t* bdst = b_s + rl * 256 + seg;
#pragma unroll
        for (int i = 0; i < 64; i += 4) {
            float4 fa = *(const float4*)(asrc + i);
            adst[i + 0] = (half_t)fa.x; adst[i + 1] = (half_t)fa.y;
            adst[i + 2] = (half_t)fa.z; adst[i + 3] = (half_t)fa.w;
            float4 fb = *(const float4*)(bsrc + i);
            bdst[i + 0] = (half_t)fb.x; bdst[i + 1] = (half_t)fb.y;
            bdst[i + 2] = (half_t)fb.z; bdst[i + 3] = (half_t)fb.w;
        }
    }
    __syncthreads();

    const int wave = tid >> 5, lane = tid & 31;
    const int nl  = lane & 15;
    const int grp = lane >> 4;
    const int kbA = grp * 16, kbB = grp * 16, mb = grp * 8;
    const int mt  = wave & 3;            // 4 row sub-tiles
    const int ntb = (wave >> 2) * 2;     // 2 col sub-tiles per wave

    v8f acc0 = {}; v8f acc1 = {};
    const int am = (mt * 16 + nl) * 256;
#pragma unroll
    for (int kk = 0; kk < 256; kk += 32) {
        v16h af;
#pragma unroll
        for (int h = 0; h < 16; ++h) af[h] = a_s[am + kk + ka_idx(kbA, h)];
        v16h bf0 = *(const v16h*)&b_s[(ntb * 16 + nl) * 256 + kk + kbB];
        v16h bf1 = *(const v16h*)&b_s[((ntb + 1) * 16 + nl) * 256 + kk + kbB];
        acc0 = wmma_f16(af, bf0, acc0);
        acc1 = wmma_f16(af, bf1, acc1);
    }
#pragma unroll
    for (int t = 0; t < 2; ++t) {
        v8f acc = t ? acc1 : acc0;
        const int cg   = c0 + (ntb + t) * 16 + nl;
        const int part = cg >> 8;               // 0=q 1=k 2=v
        const int head = (cg & 255) >> 5;
        const int dh   = cg & 31;
        half_t* dst = (part == 0) ? qw : (part == 1) ? kw : vw;
        const float sc = (part == 0) ? QK_SCALE : 1.0f;
#pragma unroll
        for (int r = 0; r < 8; ++r) {
            const int rg = m0 + mt * 16 + mb + r;
            const int b  = rg / NTOK;
            const int n  = rg - b * NTOK;
            dst[((size_t)(b * HEADS + head) * NTOK + n) * DHEAD + dh] = (half_t)(acc[r] * sc);
        }
    }
}

// ---------------------------------------------------------------------------
// Kernel 2: fused attention per (window, head). Flash-style online softmax.
// grid 2048, block 256 (8 waves; wave owns i-tiles it, it+8, ...).
// ---------------------------------------------------------------------------
__global__ __launch_bounds__(256) void attn_kernel(const half_t* __restrict__ qw,
                                                   const half_t* __restrict__ kw,
                                                   const half_t* __restrict__ vw,
                                                   const float* __restrict__ bias_table,
                                                   half_t* __restrict__ aout) {
    __shared__ alignas(32) half_t k_s[NTOK * DHEAD];
    __shared__ alignas(32) half_t v_s[NTOK * DHEAD];
    __shared__ float bias_s[1859];
    __shared__ alignas(32) half_t p_s[8][16 * 32];
    __shared__ int tdec[NTOK];

    const int tid  = threadIdx.x;
    const int b    = blockIdx.x >> 3;
    const int head = blockIdx.x & 7;
    const size_t base = (size_t)(b * HEADS + head) * NTOK * DHEAD;

    {   // K, V -> LDS (1176 x 16B each)
        const uint4* ks = (const uint4*)(kw + base);
        const uint4* vs = (const uint4*)(vw + base);
        uint4* kd = (uint4*)k_s;
        uint4* vd = (uint4*)v_s;
        for (int i = tid; i < 1176; i += 256) { kd[i] = ks[i]; vd[i] = vs[i]; }
    }
    for (int i = tid; i < 1859; i += 256) bias_s[i] = bias_table[i * HEADS + head];
    for (int i = tid; i < NTOK; i += 256) {
        int l = i / 49, s = i - l * 49, w1 = s / 7, w2 = s - w1 * 7;
        tdec[i] = (l << 8) | (w1 << 4) | w2;
    }
    __syncthreads();

    const int wave = tid >> 5, lane = tid & 31;
    const int nl  = lane & 15;
    const int grp = lane >> 4;
    const int kbA = grp * 16, kbB = grp * 16, mb = grp * 8;
    const half_t* qb = qw + base;

    for (int it = wave; it < 19; it += 8) {
        // q A-fragment for this 16-row tile (K = dh = 32, one fragment)
        const int mq = imin(it * 16 + nl, NTOK - 1);
        v16h qf;
#pragma unroll
        for (int h = 0; h < 16; ++h) qf[h] = qb[mq * DHEAD + ka_idx(kbA, h)];

        float mi[8], li[8];
        v8f o0 = {}; v8f o1 = {};
#pragma unroll
        for (int r = 0; r < 8; ++r) { mi[r] = NEG_BIG; li[r] = 0.0f; }

        for (int jc = 0; jc < 10; ++jc) {         // j in chunks of 32
            const int j0  = jc * 32;
            const int jt0 = j0 + nl, jt1 = j0 + 16 + nl;
            const int t0  = imin(jt0, NTOK - 1);
            const int t1  = imin(jt1, NTOK - 1);
            v16h kf0 = *(const v16h*)&k_s[t0 * DHEAD + kbB];
            v16h kf1 = *(const v16h*)&k_s[t1 * DHEAD + kbB];
            v8f s0 = {}; v8f s1 = {};
            s0 = wmma_f16(qf, kf0, s0);
            s1 = wmma_f16(qf, kf1, s1);

            const bool v0 = jt0 < NTOK, v1 = jt1 < NTOK;
            const int jd0 = tdec[t0], jd1 = tdec[t1];
#pragma unroll
            for (int r = 0; r < 8; ++r) {
                const int itok = it * 16 + mb + r;
                if (itok < NTOK) {
                    const int id = tdec[itok];
                    const int ia = id >> 8, i1 = (id >> 4) & 15, i2 = id & 15;
                    if (v0) {
                        int idx = (((ia - (jd0 >> 8)) + 5) * 13 + ((i1 - ((jd0 >> 4) & 15)) + 6)) * 13
                                  + ((i2 - (jd0 & 15)) + 6);
                        s0[r] += bias_s[idx];
                    }
                    if (v1) {
                        int idx = (((ia - (jd1 >> 8)) + 5) * 13 + ((i1 - ((jd1 >> 4) & 15)) + 6)) * 13
                                  + ((i2 - (jd1 & 15)) + 6);
                        s1[r] += bias_s[idx];
                    }
                }
                if (!v0) s0[r] = NEG_BIG;
                if (!v1) s1[r] = NEG_BIG;
            }

            // online softmax: rows live in 16-lane groups -> xor masks 1,2,4,8
#pragma unroll
            for (int r = 0; r < 8; ++r) {
                float cm = fmaxf(s0[r], s1[r]);
                cm = fmaxf(cm, __shfl_xor(cm, 1));
                cm = fmaxf(cm, __shfl_xor(cm, 2));
                cm = fmaxf(cm, __shfl_xor(cm, 4));
                cm = fmaxf(cm, __shfl_xor(cm, 8));
                const float mn    = fmaxf(mi[r], cm);
                const float alpha = __expf(mi[r] - mn);
                const float p0 = __expf(s0[r] - mn);
                const float p1 = __expf(s1[r] - mn);
                float rs = p0 + p1;
                rs += __shfl_xor(rs, 1);
                rs += __shfl_xor(rs, 2);
                rs += __shfl_xor(rs, 4);
                rs += __shfl_xor(rs, 8);
                li[r] = li[r] * alpha + rs;
                mi[r] = mn;
                o0[r] *= alpha;
                o1[r] *= alpha;
                p_s[wave][(mb + r) * 32 + nl]      = (half_t)p0;   // C-layout -> [m][j]
                p_s[wave][(mb + r) * 32 + 16 + nl] = (half_t)p1;
            }
            asm volatile("s_wait_dscnt 0" ::: "memory");   // wave-local LDS transpose fence

            v16h pf;
#pragma unroll
            for (int h = 0; h < 16; ++h) pf[h] = p_s[wave][nl * 32 + ka_idx(kbA, h)];
            v16h vf0, vf1;
#pragma unroll
            for (int h = 0; h < 16; ++h) {
                const int tv = imin(j0 + kbB + h, NTOK - 1);   // P==0 masks OOB rows
                vf0[h] = v_s[tv * DHEAD + nl];
                vf1[h] = v_s[tv * DHEAD + 16 + nl];
            }
            o0 = wmma_f16(pf, vf0, o0);
            o1 = wmma_f16(pf, vf1, o1);
        }

        // write normalized output: aout[(win*294 + tok)*256 + head*32 + dh], f16
#pragma unroll
        for (int r = 0; r < 8; ++r) {
            const int itok = it * 16 + mb + r;
            if (itok < NTOK) {
                const float inv = 1.0f / li[r];
                const size_t o = (size_t)(b * NTOK + itok) * DIM + head * DHEAD;
                aout[o + nl]      = (half_t)(o0[r] * inv);
                aout[o + 16 + nl] = (half_t)(o1[r] * inv);
            }
        }
    }
}

// ---------------------------------------------------------------------------
// Kernel 3: output projection, M=75264 x N=256 x K=256, f32 scatter epilogue
// into reference layout (same flat layout as input x). grid (1176, 4).
// ---------------------------------------------------------------------------
__global__ __launch_bounds__(256) void proj_kernel(const half_t* __restrict__ a,
                                                   const float* __restrict__ wout,
                                                   float* __restrict__ out) {
    __shared__ alignas(32) half_t a_s[64 * 256];
    __shared__ alignas(32) half_t b_s[64 * 256];
    const int tid = threadIdx.x;
    const int m0 = blockIdx.x * 64;
    const int c0 = blockIdx.y * 64;
    {
        const int rl  = tid >> 2;
        const int seg = (tid & 3) * 64;
        const uint4* asrc = (const uint4*)(a + (size_t)(m0 + rl) * 256 + seg);
        uint4* adst = (uint4*)(a_s + rl * 256 + seg);
#pragma unroll
        for (int i = 0; i < 8; ++i) adst[i] = asrc[i];
        const float* bsrc = wout + (size_t)(c0 + rl) * 256 + seg;
        half_t* bdst = b_s + rl * 256 + seg;
#pragma unroll
        for (int i = 0; i < 64; i += 4) {
            float4 f = *(const float4*)(bsrc + i);
            bdst[i + 0] = (half_t)f.x; bdst[i + 1] = (half_t)f.y;
            bdst[i + 2] = (half_t)f.z; bdst[i + 3] = (half_t)f.w;
        }
    }
    __syncthreads();

    const int wave = tid >> 5, lane = tid & 31;
    const int nl  = lane & 15;
    const int grp = lane >> 4;
    const int kbA = grp * 16, kbB = grp * 16, mb = grp * 8;
    const int mt  = wave & 3;
    const int ntb = (wave >> 2) * 2;

    v8f acc0 = {}; v8f acc1 = {};
    const int am = (mt * 16 + nl) * 256;
#pragma unroll
    for (int kk = 0; kk < 256; kk += 32) {
        v16h af;
#pragma unroll
        for (int h = 0; h < 16; ++h) af[h] = a_s[am + kk + ka_idx(kbA, h)];
        v16h bf0 = *(const v16h*)&b_s[(ntb * 16 + nl) * 256 + kk + kbB];
        v16h bf1 = *(const v16h*)&b_s[((ntb + 1) * 16 + nl) * 256 + kk + kbB];
        acc0 = wmma_f16(af, bf0, acc0);
        acc1 = wmma_f16(af, bf1, acc1);
    }
#pragma unroll
    for (int t = 0; t < 2; ++t) {
        v8f acc = t ? acc1 : acc0;
        const int cg = c0 + (ntb + t) * 16 + nl;
#pragma unroll
        for (int r = 0; r < 8; ++r) {
            const int rg = m0 + mt * 16 + mb + r;
            const int b  = rg / NTOK;
            const int n  = rg - b * NTOK;
            const int l  = n / 49;
            const int s  = n - l * 49;
            out[(size_t)((l * 256 + b) * 49 + s) * 256 + cg] = acc[r];
        }
    }
}

extern "C" void kernel_launch(void* const* d_in, const int* in_sizes, int n_in,
                              void* d_out, int out_size, void* d_ws, size_t ws_size,
                              hipStream_t stream) {
    const float* x          = (const float*)d_in[0];   // [1,6,16,16,7,7,256]
    const float* w_qkv      = (const float*)d_in[1];   // [768,256]
    const float* w_out      = (const float*)d_in[2];   // [256,256]
    const float* bias_table = (const float*)d_in[3];   // [1859,8]
    float* out = (float*)d_out;

    // workspace: 4 x 19,267,584 f16 tensors (q, k, v, attn_out) = ~147 MB
    const size_t TENS = (size_t)MTOT * DIM;            // 19,267,584 elements
    half_t* q_ws = (half_t*)d_ws;
    half_t* k_ws = q_ws + TENS;
    half_t* v_ws = k_ws + TENS;
    half_t* a_ws = v_ws + TENS;

    qkv_kernel<<<dim3(MTOT / 64, (3 * DIM) / 64), 256, 0, stream>>>(x, w_qkv, q_ws, k_ws, v_ws);
    attn_kernel<<<NWIN * HEADS, 256, 0, stream>>>(q_ws, k_ws, v_ws, bias_table, a_ws);
    proj_kernel<<<dim3(MTOT / 64, DIM / 64), 256, 0, stream>>>(a_ws, w_out, out);
}
